// LSTM_Encoder_83897891160515
// MI455X (gfx1250) — matrix operand order repaired and checked by reading, not compiled, not run on hardware
//
#include <hip/hip_runtime.h>
#include <hip/hip_bf16.h>

// ---------------------------------------------------------------------------
// Deep LSTM encoder for MI455X (gfx1250, wave32, WMMA).
//
//  * x / Wx / Wh / W_lin converted to bf16 once; weights pre-transposed to
//    [N][K] so each lane's WMMA B-fragment is one contiguous 32B load.
//    bf16 weights (64MB) stay resident in the 192MB L2 for the whole scan.
//  * One persistent kernel runs the [T=256] x [DEEP=4] scan as a wavefront
//    over diagonals d = t + layer: 259 device-wide barriers total.
//    256 blocks (4 layers x 64 N-tiles) x 8 wave32s. Waves split the K
//    reduction in half (w<4: K[0,512), w>=4: K[512,1024)), combine partial
//    accumulators through LDS, then waves 0-3 do the LSTM cell update fully
//    in registers (fp32 state). This halves the serial WMMA chain per
//    diagonal, which is the latency-critical resource of the recurrence.
//  * Final encoder GEMM [64 x 8192] @ [8192 x 512] also via bf16 WMMA.
// ---------------------------------------------------------------------------

typedef __bf16 bf16_t;
typedef bf16_t v16bf __attribute__((ext_vector_type(16)));
typedef bf16_t v8bf  __attribute__((ext_vector_type(8)));
typedef float  v8f   __attribute__((ext_vector_type(8)));

#define DEEP_L   4
#define BATCH    64
#define T_STEPS  256
#define EDIM     1024
#define HDIM     1024
#define GDIM     4096              // 4*H
#define ENCD     512
#define KLIN     8192              // 2*DEEP*H
#define NBLK_N   64                // HDIM / 16
#define LSTM_BLOCKS (DEEP_L * NBLK_N)   // 256
#define TPB_SCAN 256               // 8 wave32s (2 K-halves x 4 M-tiles)
#define TPB_GEMM 128               // 4 wave32s
#define KHALF    (EDIM / 2)        // 512

// ---------------- WMMA helpers (CDNA5 16x16x32 bf16, f32 accum) ------------

__device__ __forceinline__ v8f wmma_bf16(v16bf a, v16bf b, v8f c) {
  // (neg_a, A, neg_b, B, c_mod, C, reuse_a, reuse_b)
  return __builtin_amdgcn_wmma_f32_16x16x32_bf16(false, a, false, b,
                                                 (short)0, c, false, false);
}

// A-fragment, 16x32 bf16 (M x K), row-major source [rows][ld].
// Lane L: M = L%16; K = [k0 + 16*(L/16), +16) (matches B-fragment order).
__device__ __forceinline__ v16bf load_a_frag(const bf16_t* __restrict__ A,
                                             int ld, int m0, int k0, int lane) {
  const int m  = m0 + (lane & 15);
  const int kb = k0 + ((lane >> 4) << 4);
  const bf16_t* p = A + (size_t)m * ld + kb;
  return *reinterpret_cast<const v16bf*>(p);
}

// B-fragment, 32x16 bf16 (K x N), from pre-transposed weights WT[N][K].
// Lane L: N = L%16; K = [k0 + 16*(L/16), +16) -> one contiguous 32B load.
__device__ __forceinline__ v16bf load_b_frag(const bf16_t* __restrict__ WT,
                                             int ldk, int nrow0, int k0, int lane) {
  const int n  = nrow0 + (lane & 15);
  const int kb = k0 + ((lane >> 4) << 4);
  return *reinterpret_cast<const v16bf*>(WT + (size_t)n * ldk + kb);
}

__device__ __forceinline__ float fsig(float x) {
  return 1.0f / (1.0f + __expf(-x));
}
__device__ __forceinline__ float ftanh(float x) {
  return 1.0f - 2.0f / (__expf(2.0f * x) + 1.0f);
}

// Monotonic-ticket grid barrier: counter never resets -> replay-safe as long
// as it is zeroed per launch (we memset it in kernel_launch).
__device__ __forceinline__ void grid_barrier(unsigned int* cnt, unsigned int nblocks) {
  __syncthreads();
  if (threadIdx.x == 0) {
    __threadfence();
    unsigned int ticket = __hip_atomic_fetch_add(cnt, 1u, __ATOMIC_ACQ_REL,
                                                 __HIP_MEMORY_SCOPE_AGENT);
    unsigned int target = (ticket / nblocks + 1u) * nblocks;
    while (__hip_atomic_load(cnt, __ATOMIC_ACQUIRE, __HIP_MEMORY_SCOPE_AGENT) < target) {
      __builtin_amdgcn_s_sleep(1);
    }
    __threadfence();
  }
  __syncthreads();
}

// ---------------- conversion / packing kernels -----------------------------

// src [nmat][rows][cols] fp32 -> dst [nmat][cols][rows] bf16 (transpose)
__global__ void transpose_to_bf16_kernel(const float* __restrict__ src,
                                         bf16_t* __restrict__ dst,
                                         int rows, int cols, int nmat) {
  size_t i = (size_t)blockIdx.x * blockDim.x + threadIdx.x;
  size_t total = (size_t)nmat * rows * cols;
  if (i >= total) return;
  int c = (int)(i % cols);
  size_t tmp = i / cols;
  int r = (int)(tmp % rows);
  int mat = (int)(tmp / rows);
  dst[((size_t)mat * cols + c) * rows + r] = (bf16_t)src[i];
}

// x [B][T][E] fp32 -> xb [T][B][E] bf16
__global__ void pack_x_kernel(const float* __restrict__ x, bf16_t* __restrict__ xb) {
  size_t i = (size_t)blockIdx.x * blockDim.x + threadIdx.x;
  size_t total = (size_t)BATCH * T_STEPS * EDIM;
  if (i >= total) return;
  int e = (int)(i % EDIM);
  size_t tmp = i / EDIM;
  int t = (int)(tmp % T_STEPS);
  int b = (int)(tmp / T_STEPS);
  xb[((size_t)t * BATCH + b) * EDIM + e] = (bf16_t)x[i];
}

// alin [B][2*DEEP*H] bf16 with feature layout [c0|h0|c1|h1|...]
__global__ void pack_alin_kernel(const float* __restrict__ cbuf,
                                 const float* __restrict__ hfin,
                                 bf16_t* __restrict__ alin) {
  size_t i = (size_t)blockIdx.x * blockDim.x + threadIdx.x;
  size_t total = (size_t)BATCH * KLIN;
  if (i >= total) return;
  int col = (int)(i % KLIN);
  int b   = (int)(i / KLIN);
  int l = col / (2 * HDIM);
  int r = col % (2 * HDIM);
  const float* src = (r < HDIM) ? cbuf : hfin;
  int j = (r < HDIM) ? r : (r - HDIM);
  alin[i] = (bf16_t)src[((size_t)l * BATCH + b) * HDIM + j];
}

// ---------------- persistent LSTM wavefront scan ---------------------------

__global__ __launch_bounds__(TPB_SCAN) void lstm_scan_kernel(
    const bf16_t* __restrict__ xb,     // [T][B][E]
    const bf16_t* __restrict__ WxT,    // [DEEP][4H][E]   (N-major)
    const bf16_t* __restrict__ WhT,    // [DEEP][4H][H]   (N-major)
    const float*  __restrict__ bias,   // [DEEP][4H]
    bf16_t* __restrict__ hbuf,         // [2][DEEP][B][H] (t-parity double buffer)
    float*  __restrict__ cbuf,         // [DEEP][B][H]
    float*  __restrict__ hfin,         // [DEEP][B][H]    (h at t = T-1, fp32)
    unsigned int* __restrict__ barrier_cnt) {
  const int lane  = threadIdx.x & 31;
  const int wave  = threadIdx.x >> 5;   // 0..7
  const int mtile = wave & 3;           // M-tile owner
  const int khalf = wave >> 2;          // K-split half
  const int layer = blockIdx.x >> 6;    // 0..3
  const int nblk  = blockIdx.x & 63;    // 0..63
  const int n0 = nblk * 16;
  const int m0 = mtile * 16;
  const int kbeg = khalf * KHALF;
  const int kend = kbeg + KHALF;

  // partial-accumulator exchange: [4 waves][4 gates][8 rows][32 lanes] fp32
  __shared__ float lds_red[4 * 4 * 8 * 32];

  const bf16_t* Wx_l = WxT + (size_t)layer * GDIM * EDIM;
  const bf16_t* Wh_l = WhT + (size_t)layer * GDIM * HDIM;
  float* c_l = cbuf + (size_t)layer * BATCH * HDIM;

  // per-lane bias for this block's H-column (same column for every acc row)
  float bv[4];
#pragma unroll
  for (int g = 0; g < 4; ++g)
    bv[g] = bias[(size_t)layer * GDIM + g * HDIM + n0 + (lane & 15)];

  for (int d = 0; d < T_STEPS + DEEP_L - 1; ++d) {
    const int t = d - layer;            // uniform across the block
    if (t >= 0 && t < T_STEPS) {
      const int pw = t & 1;             // slot written at time t
      const int pr = pw ^ 1;            // slot holding h at time t-1
      const bf16_t* inp = (layer == 0)
          ? (xb + (size_t)t * BATCH * EDIM)
          : (hbuf + ((size_t)pw * DEEP_L + (layer - 1)) * (size_t)BATCH * HDIM);
      const bf16_t* hp =
          hbuf + ((size_t)pr * DEEP_L + layer) * (size_t)BATCH * HDIM;

      v8f acc[4] = {};   // f, i, o, g accumulators (16x16 f32 tiles)

      for (int k0 = kbeg; k0 < kend; k0 += 32) {
        v16bf a  = load_a_frag(inp, EDIM, m0, k0, lane);
        v16bf ah = load_a_frag(hp,  HDIM, m0, k0, lane);
#pragma unroll
        for (int g = 0; g < 4; ++g) {
          v16bf bx = load_b_frag(Wx_l, EDIM, g * HDIM + n0, k0, lane);
          acc[g] = wmma_bf16(a, bx, acc[g]);
          v16bf bh = load_b_frag(Wh_l, HDIM, g * HDIM + n0, k0, lane);
          acc[g] = wmma_bf16(ah, bh, acc[g]);
        }
      }

      // combine the two K-halves through LDS, then update cell state
      if (khalf == 1) {
#pragma unroll
        for (int g = 0; g < 4; ++g)
#pragma unroll
          for (int r = 0; r < 8; ++r)
            lds_red[((mtile * 4 + g) * 8 + r) * 32 + lane] = acc[g][r];
      }
      __syncthreads();
      if (khalf == 0) {
#pragma unroll
        for (int g = 0; g < 4; ++g)
#pragma unroll
          for (int r = 0; r < 8; ++r)
            acc[g][r] += lds_red[((mtile * 4 + g) * 8 + r) * 32 + lane];

        bf16_t* hw =
            hbuf + ((size_t)pw * DEEP_L + layer) * (size_t)BATCH * HDIM;
        const int n = n0 + (lane & 15);
#pragma unroll
        for (int r = 0; r < 8; ++r) {
          const int m = m0 + r + ((lane >> 4) << 3);  // C/D layout: M=r+8*(L/16)
          const size_t idx = (size_t)m * HDIM + n;
          float fg = fsig(acc[0][r] + bv[0]);
          float ig = fsig(acc[1][r] + bv[1]);
          float og = fsig(acc[2][r] + bv[2]);
          float gg = ftanh(acc[3][r] + bv[3]);
          float cn = c_l[idx] * fg + gg * ig;
          float hn = og * ftanh(cn);
          c_l[idx] = cn;
          hw[idx]  = (bf16_t)hn;
          if (t == T_STEPS - 1)
            hfin[(size_t)layer * BATCH * HDIM + idx] = hn;
        }
      }
    }
    // also orders LDS reuse across diagonals (contains __syncthreads)
    grid_barrier(barrier_cnt, LSTM_BLOCKS);
  }
}

// ---------------- final encoder GEMM: [64 x 8192] @ [8192 x 512] -----------

__global__ __launch_bounds__(TPB_GEMM) void final_gemm_kernel(
    const bf16_t* __restrict__ A,    // [64][8192] bf16
    const bf16_t* __restrict__ BT,   // [512][8192] bf16 (N-major)
    const float*  __restrict__ blin, // [512]
    float* __restrict__ out) {       // [64][512]
  const int lane = threadIdx.x & 31;
  const int wave = threadIdx.x >> 5;
  const int n0 = blockIdx.x * 16;
  const int m0 = wave * 16;

  v8f acc = {};
  for (int k0 = 0; k0 < KLIN; k0 += 32) {
    v16bf a = load_a_frag(A, KLIN, m0, k0, lane);
    v16bf b = load_b_frag(BT, KLIN, n0, k0, lane);
    acc = wmma_bf16(a, b, acc);
  }
  const int n = n0 + (lane & 15);
  const float bv = blin[n];
#pragma unroll
  for (int r = 0; r < 8; ++r) {
    const int m = m0 + r + ((lane >> 4) << 3);
    out[(size_t)m * ENCD + n] = acc[r] + bv;
  }
}

// ---------------------------------------------------------------------------

extern "C" void kernel_launch(void* const* d_in, const int* in_sizes, int n_in,
                              void* d_out, int out_size, void* d_ws, size_t ws_size,
                              hipStream_t stream) {
  const float* x    = (const float*)d_in[0];  // [B,T,E]
  const float* Wx   = (const float*)d_in[1];  // [DEEP,E,4H]
  const float* Wh   = (const float*)d_in[2];  // [DEEP,H,4H]
  const float* bias = (const float*)d_in[3];  // [DEEP,4H]
  const float* Wlin = (const float*)d_in[4];  // [8192,512]
  const float* blin = (const float*)d_in[5];  // [512]
  float* out = (float*)d_out;

  char* ws = (char*)d_ws;
  size_t off = 0;
  auto carve = [&](size_t bytes) -> void* {
    off = (off + 255) & ~(size_t)255;
    void* p = ws + off;
    off += bytes;
    return p;
  };

  const size_t wxt_b  = (size_t)DEEP_L * GDIM * EDIM * sizeof(bf16_t);   // 32MB
  const size_t wht_b  = (size_t)DEEP_L * GDIM * HDIM * sizeof(bf16_t);   // 32MB
  const size_t xb_b   = (size_t)T_STEPS * BATCH * EDIM * sizeof(bf16_t); // 32MB
  const size_t hbuf_b = (size_t)2 * DEEP_L * BATCH * HDIM * sizeof(bf16_t);
  const size_t cbuf_b = (size_t)DEEP_L * BATCH * HDIM * sizeof(float);
  const size_t hfin_b = (size_t)DEEP_L * BATCH * HDIM * sizeof(float);
  const size_t alin_b = (size_t)BATCH * KLIN * sizeof(bf16_t);
  const size_t wlt_b  = (size_t)ENCD * KLIN * sizeof(bf16_t);            // 8MB

  bf16_t* WxT  = (bf16_t*)carve(wxt_b);
  bf16_t* WhT  = (bf16_t*)carve(wht_b);
  bf16_t* xb   = (bf16_t*)carve(xb_b);
  bf16_t* hbuf = (bf16_t*)carve(hbuf_b);
  float*  cbuf = (float*)carve(cbuf_b);
  float*  hfin = (float*)carve(hfin_b);
  bf16_t* alin = (bf16_t*)carve(alin_b);
  bf16_t* WlT  = (bf16_t*)carve(wlt_b);
  unsigned int* bar = (unsigned int*)carve(256);

  // per-launch init (graph-capture legal)
  hipMemsetAsync(hbuf, 0, hbuf_b, stream);
  hipMemsetAsync(cbuf, 0, cbuf_b, stream);
  hipMemsetAsync(bar,  0, 256,    stream);

  const int tpb = 256;
  // Wx: [DEEP][E][4H] -> [DEEP][4H][E] bf16
  {
    size_t total = (size_t)DEEP_L * EDIM * GDIM;
    transpose_to_bf16_kernel<<<(unsigned)((total + tpb - 1) / tpb), tpb, 0, stream>>>(
        Wx, WxT, EDIM, GDIM, DEEP_L);
  }
  // Wh: [DEEP][H][4H] -> [DEEP][4H][H] bf16
  {
    size_t total = (size_t)DEEP_L * HDIM * GDIM;
    transpose_to_bf16_kernel<<<(unsigned)((total + tpb - 1) / tpb), tpb, 0, stream>>>(
        Wh, WhT, HDIM, GDIM, DEEP_L);
  }
  // W_lin: [8192][512] -> [512][8192] bf16
  {
    size_t total = (size_t)KLIN * ENCD;
    transpose_to_bf16_kernel<<<(unsigned)((total + tpb - 1) / tpb), tpb, 0, stream>>>(
        Wlin, WlT, KLIN, ENCD, 1);
  }
  // x -> [T][B][E] bf16
  {
    size_t total = (size_t)BATCH * T_STEPS * EDIM;
    pack_x_kernel<<<(unsigned)((total + tpb - 1) / tpb), tpb, 0, stream>>>(x, xb);
  }

  // persistent wavefront scan: 256 co-resident blocks, 8 wave32s each
  lstm_scan_kernel<<<LSTM_BLOCKS, TPB_SCAN, 0, stream>>>(
      xb, WxT, WhT, bias, hbuf, cbuf, hfin, bar);

  // pack [c0|h0|c1|h1|...] and run the encoder GEMM
  {
    size_t total = (size_t)BATCH * KLIN;
    pack_alin_kernel<<<(unsigned)((total + tpb - 1) / tpb), tpb, 0, stream>>>(
        cbuf, hfin, alin);
  }
  final_gemm_kernel<<<ENCD / 16, TPB_GEMM, 0, stream>>>(alin, WlT, blin, out);
}
